// LRU_4776003633532
// MI455X (gfx1250) — matrix operand order repaired and checked by reading, not compiled, not run on hardware
//
#include <hip/hip_runtime.h>
#include <hip/hip_bf16.h>
#include <math.h>

#define D 256
#define BATCH 8
#define SEQ 4096
#define M_TOT (BATCH * SEQ)   // 32768
#define CHUNK 64
#define NCHUNK (SEQ / CHUNK)  // 64

// sched_barrier mask: allow ALU/VALU/SALU/WMMA/trans to cross, pin VMEM & DS.
#define SCHED_PIN_MEM 0x40F
// Fence that pins memory ops at IR/DAG level too (WMMA & ALU may still cross).
#define MEM_FENCE() do { asm volatile("" ::: "memory"); \
                         __builtin_amdgcn_sched_barrier(SCHED_PIN_MEM); } while (0)
// Value pin: per-fragment wait point right before its consumer.
#define PIN_FRAG(f) asm volatile("" : "+v"(f))

typedef __attribute__((ext_vector_type(16))) __bf16 v16bf;
typedef __attribute__((ext_vector_type(8)))  __bf16 v8bf;
typedef __attribute__((ext_vector_type(4)))  __bf16 v4bf;
typedef __attribute__((ext_vector_type(8)))  float  v8f;

union BFrag { v16bf v; v8bf h[2]; };

// ---------------------------------------------------------------------------
// Kernel A: parameter prep.
// ---------------------------------------------------------------------------
__global__ void lru_prep(const float* __restrict__ B_re, const float* __restrict__ B_im,
                         const float* __restrict__ C_re, const float* __restrict__ C_im,
                         const float* __restrict__ nu_log, const float* __restrict__ theta_log,
                         const float* __restrict__ gamma_log,
                         __bf16* __restrict__ Wre, __bf16* __restrict__ Wim,
                         __bf16* __restrict__ Cre16, __bf16* __restrict__ CimN16,
                         float* __restrict__ lre, float* __restrict__ lim) {
    const int idx = blockIdx.x * 256 + threadIdx.x;   // [0, 65536)
    const int row = idx >> 8;
    const float g = expf(gamma_log[row]);
    Wre[idx]    = (__bf16)(B_re[idx] * g);
    Wim[idx]    = (__bf16)(B_im[idx] * g);
    Cre16[idx]  = (__bf16)(C_re[idx]);
    CimN16[idx] = (__bf16)(-C_im[idx]);
    if (idx < D) {
        const float mod = expf(-expf(nu_log[idx]));
        const float th  = expf(theta_log[idx]);
        lre[idx] = mod * cosf(th);
        lim[idx] = mod * sinf(th);
    }
}

// ---------------------------------------------------------------------------
// Kernel B: input projection  u_{re,im} = x @ W_{re,im}^T  via bf16 WMMA.
// Per K-step: all 8 B-fragments issued as one clause (pinned above the WMMA
// group by the memory fence; loads are not invariant since the streamed
// operands carry no __restrict__). Each fragment is value-pinned immediately
// before its WMMA so the s_wait_loadcnt stagger across the burst.
// Wave tile: 16M x 128N, one matrix (re|im). Workgroup: 8 waves. Grid: M/32.
// ---------------------------------------------------------------------------
__global__ void __launch_bounds__(256)
__attribute__((amdgpu_waves_per_eu(1)))
lru_in_gemm(const float* __restrict__ x,
            const __bf16* Wre, const __bf16* Wim,
            float* __restrict__ ure, float* __restrict__ uim) {
    __shared__ __bf16 xlds[32 * D];   // 16 KB
    const int tid = threadIdx.x;

    // Warm L2 for the weight matrices while we stage x (global_prefetch_b8).
    {
        const char* pre = (const char*)Wre;   // Wre and Wim contiguous in ws
        for (int i = tid; i < 2048; i += 256) // 2 * 128 KB / 128 B lines
            __builtin_prefetch(pre + (size_t)i * 128, 0, 0);
    }

    // Stage 32 x D f32 -> bf16 into LDS (vectorized float4 reads).
    const float* xg = x + (size_t)blockIdx.x * 32 * D;
    for (int i = tid; i < 32 * D / 4; i += 256) {
        float4 f = ((const float4*)xg)[i];
        v4bf bv;
        bv[0] = (__bf16)f.x; bv[1] = (__bf16)f.y;
        bv[2] = (__bf16)f.z; bv[3] = (__bf16)f.w;
        *(v4bf*)&xlds[i * 4] = bv;
    }
    __syncthreads();

    const int w     = tid >> 5;
    const int lane  = tid & 31;
    const int mat   = w & 1;            // 0 = re, 1 = im
    const int nhalf = (w >> 1) & 1;     // 0..1 -> 128-col half
    const int msub  = w >> 2;           // 0..1 -> 16-row sub-tile
    const int l15   = lane & 15;
    const int hi    = (lane >> 4) & 1;

    v8f acc[8];
#pragma unroll
    for (int i = 0; i < 8; ++i) acc[i] = (v8f)0.0f;

    const __bf16* W     = mat ? Wim : Wre;
    const __bf16* xrow  = &xlds[(msub * 16 + l15) * D];
    const __bf16* wbase = W + (size_t)(nhalf * 128 + l15) * D + hi * 16;

    auto lda = [&](BFrag& f, int kk) {
        const int ka = kk * 32 + hi * 16;
        f.h[0] = *(const v8bf*)(xrow + ka);
        f.h[1] = *(const v8bf*)(xrow + ka + 8);
    };
    auto ldb = [&](BFrag& f, int nt, int kk) {
        const __bf16* p = wbase + (size_t)nt * 16 * D + kk * 32;
        f.h[0] = *(const v8bf*)(p);
        f.h[1] = *(const v8bf*)(p + 8);
    };

    BFrag aq[2];
    lda(aq[0], 0);

#pragma unroll
    for (int kk = 0; kk < 8; ++kk) {         // K = 256 in steps of 32
        BFrag bq[8];
#pragma unroll
        for (int nt = 0; nt < 8; ++nt) ldb(bq[nt], nt, kk);
        if (kk < 7) lda(aq[(kk + 1) & 1], kk + 1);   // ds_load for next K-step
        MEM_FENCE();
#pragma unroll
        for (int nt = 0; nt < 8; ++nt) {
            PIN_FRAG(bq[nt].v);              // per-fragment wait point
            acc[nt] = __builtin_amdgcn_wmma_f32_16x16x32_bf16(
                false, aq[kk & 1].v, false, bq[nt].v, (short)0, acc[nt],
                false, false);
        }
    }

    // C/D layout: VGPR j -> M = j + 8*hi, N = lane&15.
    float* ug = mat ? uim : ure;
    const int mrow0 = blockIdx.x * 32 + msub * 16 + hi * 8;
    const int ncol0 = nhalf * 128 + l15;
#pragma unroll
    for (int nt = 0; nt < 8; ++nt)
#pragma unroll
        for (int j = 0; j < 8; ++j)
            ug[(size_t)(mrow0 + j) * D + nt * 16 + ncol0] = acc[nt][j];
}

// ---------------------------------------------------------------------------
// Kernel C: chunk-local scans. Thread = (b, d, chunk); h starts at 0,
// stores chunk end-state to carry buffer [b][chunk][d].
// ---------------------------------------------------------------------------
__global__ void lru_scan_local(const float* __restrict__ ure, const float* __restrict__ uim,
                               const float* __restrict__ lre, const float* __restrict__ lim,
                               float* __restrict__ cre, float* __restrict__ cim) {
    const int idx   = blockIdx.x * 256 + threadIdx.x;  // [0, 131072)
    const int d     = idx & 255;
    const int chunk = (idx >> 8) & (NCHUNK - 1);
    const int b     = idx >> 14;
    const float ar = lre[d], ai = lim[d];
    float hr = 0.f, hv = 0.f;
    const size_t base = ((size_t)b * SEQ + (size_t)chunk * CHUNK) * D + d;
    for (int t = 0; t < CHUNK; ++t) {
        const size_t m = base + (size_t)t * D;
        const float nr = ar * hr - ai * hv + ure[m];
        const float ni = ar * hv + ai * hr + uim[m];
        hr = nr; hv = ni;
    }
    const size_t coff = ((size_t)b * NCHUNK + chunk) * D + d;
    cre[coff] = hr; cim[coff] = hv;
}

// ---------------------------------------------------------------------------
// Kernel D: per-(b,d) exclusive prefix over chunk carries.
// lambda^CHUNK by 6 complex squarings (CHUNK = 64 = 2^6).
// ---------------------------------------------------------------------------
__global__ void lru_scan_prefix(const float* __restrict__ lre, const float* __restrict__ lim,
                                float* __restrict__ cre, float* __restrict__ cim) {
    const int idx = blockIdx.x * 256 + threadIdx.x;    // [0, 2048)
    const int d = idx & 255, b = idx >> 8;
    float pr = lre[d], pi = lim[d];
    for (int i = 0; i < 6; ++i) {
        const float nr = pr * pr - pi * pi;
        const float ni = 2.f * pr * pi;
        pr = nr; pi = ni;
    }
    float sr = 0.f, si = 0.f;
    for (int c = 0; c < NCHUNK; ++c) {
        const size_t off = ((size_t)b * NCHUNK + c) * D + d;
        const float er = cre[off], ei = cim[off];
        cre[off] = sr; cim[off] = si;                  // exclusive prefix
        const float nr = pr * sr - pi * si + er;
        const float ni = pr * si + pi * sr + ei;
        sr = nr; si = ni;
    }
}

// ---------------------------------------------------------------------------
// Kernel E: re-scan each chunk seeded with its prefix, emit h as bf16.
// ---------------------------------------------------------------------------
__global__ void lru_scan_apply(const float* __restrict__ ure, const float* __restrict__ uim,
                               const float* __restrict__ lre, const float* __restrict__ lim,
                               const float* __restrict__ cre, const float* __restrict__ cim,
                               __bf16* __restrict__ hre, __bf16* __restrict__ him) {
    const int idx   = blockIdx.x * 256 + threadIdx.x;
    const int d     = idx & 255;
    const int chunk = (idx >> 8) & (NCHUNK - 1);
    const int b     = idx >> 14;
    const float ar = lre[d], ai = lim[d];
    const size_t coff = ((size_t)b * NCHUNK + chunk) * D + d;
    float hr = cre[coff], hv = cim[coff];
    const size_t base = ((size_t)b * SEQ + (size_t)chunk * CHUNK) * D + d;
    for (int t = 0; t < CHUNK; ++t) {
        const size_t m = base + (size_t)t * D;
        const float nr = ar * hr - ai * hv + ure[m];
        const float ni = ar * hv + ai * hr + uim[m];
        hr = nr; hv = ni;
        hre[m] = (__bf16)hr;
        him[m] = (__bf16)hv;
    }
}

// ---------------------------------------------------------------------------
// Kernel F: output projection  y = h_re @ Cre^T + h_im @ (-Cim)^T  (f32 out).
// Per K-step: two groups of 4 (Cre,CimN) fragment pairs, loads fenced above
// the group, each pair value-pinned right before its two WMMAs.
// Wave tile: 16M x 128N. Workgroup: 8 waves. Grid: M/64.
// ---------------------------------------------------------------------------
__global__ void __launch_bounds__(256)
__attribute__((amdgpu_waves_per_eu(1)))
lru_out_gemm(const __bf16* hre, const __bf16* him,
             const __bf16* Cre, const __bf16* CimN,
             float* __restrict__ y) {
    const int tid   = threadIdx.x;
    const int w     = tid >> 5;
    const int lane  = tid & 31;
    const int nhalf = w & 1;            // 128-col half
    const int msub  = w >> 1;           // 0..3 -> 16-row sub-tile
    const int l15   = lane & 15;
    const int hi    = (lane >> 4) & 1;
    const int rbase = blockIdx.x * 64 + msub * 16;

    // Warm L2 for C weights (global_prefetch_b8).
    {
        const char* pre = (const char*)Cre;   // Cre16 and CimN16 contiguous in ws
        for (int i = tid; i < 2048; i += 256)
            __builtin_prefetch(pre + (size_t)i * 128, 0, 0);
    }

    v8f acc[8];
#pragma unroll
    for (int i = 0; i < 8; ++i) acc[i] = (v8f)0.0f;

    const __bf16* hre_row = hre + (size_t)(rbase + l15) * D;
    const __bf16* him_row = him + (size_t)(rbase + l15) * D;
    const __bf16* crbase  = Cre  + (size_t)(nhalf * 128 + l15) * D + hi * 16;
    const __bf16* cibase  = CimN + (size_t)(nhalf * 128 + l15) * D + hi * 16;

    auto lda = [&](BFrag& fr, BFrag& fi, int kk) {
        const int ka = kk * 32 + hi * 16;
        fr.h[0] = *(const v8bf*)(hre_row + ka);
        fr.h[1] = *(const v8bf*)(hre_row + ka + 8);
        fi.h[0] = *(const v8bf*)(him_row + ka);
        fi.h[1] = *(const v8bf*)(him_row + ka + 8);
    };
    auto ldb = [&](BFrag& fr, BFrag& fi, int nt, int kk) {
        const __bf16* pr = crbase + (size_t)nt * 16 * D + kk * 32;
        const __bf16* pi = cibase + (size_t)nt * 16 * D + kk * 32;
        fr.h[0] = *(const v8bf*)(pr); fr.h[1] = *(const v8bf*)(pr + 8);
        fi.h[0] = *(const v8bf*)(pi); fi.h[1] = *(const v8bf*)(pi + 8);
    };

    BFrag arq[2], aiq[2];
    lda(arq[0], aiq[0], 0);

#pragma unroll
    for (int kk = 0; kk < 8; ++kk) {
        if (kk < 7) lda(arq[(kk + 1) & 1], aiq[(kk + 1) & 1], kk + 1);
#pragma unroll
        for (int g = 0; g < 2; ++g) {
            BFrag br[4], bi[4];
#pragma unroll
            for (int i = 0; i < 4; ++i) ldb(br[i], bi[i], g * 4 + i, kk);
            MEM_FENCE();
#pragma unroll
            for (int i = 0; i < 4; ++i) {
                const int nt = g * 4 + i;
                PIN_FRAG(br[i].v);
                acc[nt] = __builtin_amdgcn_wmma_f32_16x16x32_bf16(
                    false, arq[kk & 1].v, false, br[i].v, (short)0, acc[nt],
                    false, false);
                PIN_FRAG(bi[i].v);
                acc[nt] = __builtin_amdgcn_wmma_f32_16x16x32_bf16(
                    false, aiq[kk & 1].v, false, bi[i].v, (short)0, acc[nt],
                    false, false);
            }
        }
    }

    const int mrow0 = rbase + hi * 8;
    const int ncol0 = nhalf * 128 + l15;
#pragma unroll
    for (int nt = 0; nt < 8; ++nt)
#pragma unroll
        for (int j = 0; j < 8; ++j)
            y[(size_t)(mrow0 + j) * D + nt * 16 + ncol0] = acc[nt][j];
}

// ---------------------------------------------------------------------------
extern "C" void kernel_launch(void* const* d_in, const int* in_sizes, int n_in,
                              void* d_out, int out_size, void* d_ws, size_t ws_size,
                              hipStream_t stream) {
    (void)in_sizes; (void)n_in; (void)out_size; (void)ws_size;

    const float* x         = (const float*)d_in[0];
    const float* nu_log    = (const float*)d_in[1];
    const float* theta_log = (const float*)d_in[2];
    const float* gamma_log = (const float*)d_in[3];
    const float* B_re      = (const float*)d_in[4];
    const float* B_im      = (const float*)d_in[5];
    const float* C_re      = (const float*)d_in[6];
    const float* C_im      = (const float*)d_in[7];

    // Workspace carve-up (256 B aligned chunks), ~98 MB total.
    char*  ws  = (char*)d_ws;
    size_t off = 0;
    auto carve = [&](size_t bytes) -> char* {
        char* p = ws + off;
        off += (bytes + 255) & ~(size_t)255;
        return p;
    };
    __bf16* Wre    = (__bf16*)carve((size_t)D * D * 2);
    __bf16* Wim    = (__bf16*)carve((size_t)D * D * 2);
    __bf16* Cre16  = (__bf16*)carve((size_t)D * D * 2);
    __bf16* CimN16 = (__bf16*)carve((size_t)D * D * 2);
    float*  lre    = (float*) carve((size_t)D * 4);
    float*  lim    = (float*) carve((size_t)D * 4);
    float*  cre    = (float*) carve((size_t)BATCH * NCHUNK * D * 4);
    float*  cim    = (float*) carve((size_t)BATCH * NCHUNK * D * 4);
    float*  ure    = (float*) carve((size_t)M_TOT * D * 4);
    float*  uim    = (float*) carve((size_t)M_TOT * D * 4);
    __bf16* hre    = (__bf16*)carve((size_t)M_TOT * D * 2);
    __bf16* him    = (__bf16*)carve((size_t)M_TOT * D * 2);

    float* y = (float*)d_out;

    lru_prep<<<D * D / 256, 256, 0, stream>>>(B_re, B_im, C_re, C_im,
                                              nu_log, theta_log, gamma_log,
                                              Wre, Wim, Cre16, CimN16, lre, lim);
    lru_in_gemm<<<M_TOT / 32, 256, 0, stream>>>(x, Wre, Wim, ure, uim);
    lru_scan_local<<<BATCH * NCHUNK * D / 256, 256, 0, stream>>>(ure, uim, lre, lim, cre, cim);
    lru_scan_prefix<<<BATCH * D / 256, 256, 0, stream>>>(lre, lim, cre, cim);
    lru_scan_apply<<<BATCH * NCHUNK * D / 256, 256, 0, stream>>>(ure, uim, lre, lim,
                                                                 cre, cim, hre, him);
    lru_out_gemm<<<M_TOT / 64, 256, 0, stream>>>(hre, him, Cre16, CimN16, y);
}
